// StateOnlyRNN_40570261078251
// MI455X (gfx1250) — matrix operand order repaired and checked by reading, not compiled, not run on hardware
//
#include <hip/hip_runtime.h>

// ---- CDNA5 (gfx1250) wave32 WMMA types ----
typedef __attribute__((ext_vector_type(16))) _Float16 v16h;
typedef __attribute__((ext_vector_type(8)))  _Float16 v8h;
typedef __attribute__((ext_vector_type(8)))  float    v8f;
typedef __attribute__((ext_vector_type(4)))  float    f32x4;

#define HD     256   // hidden dim
#define PL     512   // pred_len
#define NB     1024  // batch
#define PITCH  264   // LDS row pitch (halves): 528B = 132 dw, 132 % 64 = 4
                     // -> 16 rows hit 16 distinct 4-dword bank windows

// Block: 256 threads = 8 wave32, owns 16 batch rows.
// Wave w owns output columns [32w, 32w+32): two 16x16 f32 C/D tiles kept in
// VGPRs as the master state; all of its W^T fragments (128 VGPRs) register-
// resident for all 512 steps -> per-step LDS traffic = A-gather only.
__global__ __launch_bounds__(256)
void rnn_wmma_kernel(const float* __restrict__ x,
                     const float* __restrict__ W,
                     const float* __restrict__ bias,
                     float* __restrict__ out)
{
  __shared__ _Float16 hbuf0[16 * PITCH];   // f16 h ping (8.25 KB)
  __shared__ _Float16 hbuf1[16 * PITCH];   // f16 h pong

  const int tid  = threadIdx.x;
  const int lane = tid & 31;
  const int wv   = tid >> 5;          // wave id 0..7
  const int lm   = lane & 15;
  const int hi   = lane >> 4;         // half-wave select (0/1)
  const int n0   = wv * 32;           // wave's first output column
  const int b0   = blockIdx.x * 16;   // block's first batch row

  // ---- one-time: W^T B-fragments straight from global f32 into registers.
  // B k order chosen here (A gather matches): lane = column n, 16 consecutive K halves per
  // lane (lanes 0-15: K 0..15 of the chunk, lanes 16-31: K 16..31) = a
  // contiguous run of row-major W row n.  No transpose, no LDS staging.
  v16h bw[16];  // [tile*8 + kc]
#pragma unroll
  for (int tile = 0; tile < 2; ++tile) {
    const float* wrow = W + (size_t)(n0 + tile * 16 + lm) * HD + hi * 16;
#pragma unroll
    for (int kc = 0; kc < 8; ++kc) {
      v16h f;
#pragma unroll
      for (int q = 0; q < 4; ++q) {
        f32x4 w4 = *(const f32x4*)(wrow + kc * 32 + q * 4);
        f[q * 4 + 0] = (_Float16)w4[0];
        f[q * 4 + 1] = (_Float16)w4[1];
        f[q * 4 + 2] = (_Float16)w4[2];
        f[q * 4 + 3] = (_Float16)w4[3];
      }
      bw[tile * 8 + kc] = f;
    }
  }

  // ---- init state accumulators from x (C/D layout: VGPR r -> row r+8*hi,
  // lane%16 -> column) ----
  v8f c0, c1;
  {
    const float* xp = x + (size_t)(b0 + hi * 8) * HD + n0 + lm;
#pragma unroll
    for (int r = 0; r < 8; ++r) {
      c0[r] = xp[(size_t)r * HD];
      c1[r] = xp[(size_t)r * HD + 16];
    }
  }

  const float bias0 = bias[n0 + lm];
  const float bias1 = bias[n0 + 16 + lm];

  // per-lane output base: out[b][n][t], b = b0+8*hi (+r), n = n0+lm (tile1 +16)
  float* outp = out + ((size_t)(b0 + hi * 8) * HD + (n0 + lm)) * PL;

  _Float16* hp0 = hbuf0 + (hi * 8) * PITCH + n0 + lm;   // publish base (ping)
  _Float16* hp1 = hbuf1 + (hi * 8) * PITCH + n0 + lm;   // publish base (pong)
  const _Float16* ap0 = hbuf0 + lm * PITCH + hi * 16;   // gather base (ping)
  const _Float16* ap1 = hbuf1 + lm * PITCH + hi * 16;   // gather base (pong)

  auto step = [&](int t, _Float16* __restrict__ hp,
                  const _Float16* __restrict__ ap) {
    // (1) publish f16 copy of this wave's 32 state columns for A assembly
#pragma unroll
    for (int r = 0; r < 8; ++r) {
      hp[r * PITCH]      = (_Float16)c0[r];
      hp[r * PITCH + 16] = (_Float16)c1[r];
    }

    // (2) split workgroup barrier: signal arrival once the publish is visible,
    //     then hide the barrier wait behind the 16 independent global emits.
    asm volatile("s_wait_dscnt 0x0\n\ts_barrier_signal -1" ::: "memory");

    // emit pre-transform state h_t (f32, contiguous in t -> L2 write-combines)
#pragma unroll
    for (int r = 0; r < 8; ++r) {
      outp[(size_t)r * (HD * PL) + t]           = c0[r];
      outp[(size_t)r * (HD * PL) + 16 * PL + t] = c1[r];
    }

    asm volatile("s_barrier_wait -1" ::: "memory");

    // (3)+(4) gather A fragments with one-chunk lookahead so the legalizer
    // can use partial s_wait_dscnt, and run two interleaved WMMA chains with
    // register-resident B.  h_{t+1} = h_t @ W^T + b, f32 accumulate.
    v16h aCur, aNxt;
    {
      v8h lo  = *(const v8h*)(ap);
      v8h hi8 = *(const v8h*)(ap + 8);
      aCur = __builtin_shufflevector(lo, hi8,
               0, 1, 2, 3, 4, 5, 6, 7, 8, 9, 10, 11, 12, 13, 14, 15);
    }
#pragma unroll
    for (int r = 0; r < 8; ++r) { c0[r] = bias0; c1[r] = bias1; }

#pragma unroll
    for (int kc = 0; kc < 8; ++kc) {
      if (kc < 7) {  // prefetch next A chunk before consuming current one
        v8h lo  = *(const v8h*)(ap + (kc + 1) * 32);
        v8h hi8 = *(const v8h*)(ap + (kc + 1) * 32 + 8);
        aNxt = __builtin_shufflevector(lo, hi8,
                 0, 1, 2, 3, 4, 5, 6, 7, 8, 9, 10, 11, 12, 13, 14, 15);
      }
      c0 = __builtin_amdgcn_wmma_f32_16x16x32_f16(false, aCur, false, bw[kc],
                                                  (short)0, c0, false, false);
      c1 = __builtin_amdgcn_wmma_f32_16x16x32_f16(false, aCur, false, bw[8 + kc],
                                                  (short)0, c1, false, false);
      aCur = aNxt;
    }
  };

#pragma clang loop unroll(disable)
  for (int t2 = 0; t2 < PL; t2 += 2) {
    step(t2,     hp0, ap0);   // static ping/pong addressing
    step(t2 + 1, hp1, ap1);
  }
}

extern "C" void kernel_launch(void* const* d_in, const int* in_sizes, int n_in,
                              void* d_out, int out_size, void* d_ws, size_t ws_size,
                              hipStream_t stream) {
  const float* x  = (const float*)d_in[0];  // (1024, 256)
  const float* W  = (const float*)d_in[1];  // (256, 256)
  const float* b  = (const float*)d_in[2];  // (256,)
  float* out      = (float*)d_out;          // (1024, 256, 512)

  rnn_wmma_kernel<<<dim3(NB / 16), dim3(256), 0, stream>>>(x, W, b, out);
}
